// AttentionModel_39848706573073
// MI455X (gfx1250) — matrix operand order repaired and checked by reading, not compiled, not run on hardware
//
#include <hip/hip_runtime.h>

#define E_DIM 1024
#define S_LEN 2048
#define TK    64      // key-block width in the attention loop

typedef __attribute__((ext_vector_type(16))) _Float16 v16h;
typedef __attribute__((ext_vector_type(8)))  _Float16 v8h;
typedef __attribute__((ext_vector_type(8)))  float    v8f;

static __device__ __forceinline__ v8f wmma_f16(v16h a, v16h b, v8f c) {
  // 8 args: (neg_a, A, neg_b, B, c_mod, C, reuse_a, reuse_b)
  return __builtin_amdgcn_wmma_f32_16x16x32_f16(false, a, false, b, (short)0, c,
                                                false, false);
}

// A-fragment (16x32 f16, row-major source). p = row base + k0 + 8*hl.
// lanes 0-15 hold K=k0+0..7 / k0+16..23 ; lanes 16-31 hold K=k0+8..15 / k0+24..31
static __device__ __forceinline__ v16h ld_afrag(const _Float16* p) {
  v8h c0 = *(const v8h*)(p);
  v8h c1 = *(const v8h*)(p + 16);
  v16h r;
#pragma unroll
  for (int i = 0; i < 8; ++i) { r[i] = c0[i]; r[i + 8] = c1[i]; }
  return r;
}

// B-fragment (32x16 f16, K x N): lane holds column N, K=k0+8*hl+0..7 / k0+16+8*hl+0..7.
// p = (N-row base in an N-major/K-contiguous layout) + k0 + 8*hl.
static __device__ __forceinline__ v16h ld_bfrag(const _Float16* p) {
  v8h c0 = *(const v8h*)(p);
  v8h c1 = *(const v8h*)(p + 16);
  v16h r;
#pragma unroll
  for (int i = 0; i < 8; ++i) { r[i] = c0[i]; r[i + 8] = c1[i]; }
  return r;
}

// Issue one per-lane async 16B copy global -> LDS (ASYNCcnt tracked).
static __device__ __forceinline__ void async_g2l_b128(uint32_t lds_off,
                                                      const void* gptr) {
  asm volatile("global_load_async_to_lds_b128 %0, %1, off"
               :: "v"(lds_off), "v"((unsigned long long)(uintptr_t)gptr)
               : "memory");
}
static __device__ __forceinline__ void wait_async0() {
  asm volatile("s_wait_asynccnt 0x0" ::: "memory");
}

// ---------------------------------------------------------------------------
// Kernel 0: bulk f32 -> f16 conversion (for the weight matrices).
// ---------------------------------------------------------------------------
__global__ __launch_bounds__(256) void cvt16_kernel(const float* __restrict__ in,
                                                    _Float16* __restrict__ out,
                                                    int n) {
  const int i = (blockIdx.x * 256 + threadIdx.x) * 8;
  if (i + 8 <= n) {
    const float4 f0 = *(const float4*)(in + i);
    const float4 f1 = *(const float4*)(in + i + 4);
    v8h o;
    o[0] = (_Float16)f0.x; o[1] = (_Float16)f0.y;
    o[2] = (_Float16)f0.z; o[3] = (_Float16)f0.w;
    o[4] = (_Float16)f1.x; o[5] = (_Float16)f1.y;
    o[6] = (_Float16)f1.z; o[7] = (_Float16)f1.w;
    *(v8h*)(out + i) = o;
  }
}

// ---------------------------------------------------------------------------
// Kernel 1: P = X @ W^T + b  (X f32, W pre-converted f16, out f16).
// One block per 16-row m-tile; X rows staged once into LDS as f16 and shared
// by all 8 waves; wave w computes columns [w*128, w*128+128).
// transposeOut==0: out is [8192][1024] f16 row-major (Q, K)
// transposeOut==1: out is [4][1024][2048] f16 (vT[b][e][t], for V)
// ---------------------------------------------------------------------------
__global__ __launch_bounds__(256) void proj_kernel(const float* __restrict__ X,
                                                   const _Float16* __restrict__ W16,
                                                   const float* __restrict__ bias,
                                                   _Float16* __restrict__ out,
                                                   int transposeOut) {
  __shared__ _Float16 xs[16][E_DIM];   // 32 KB staged activations

  const int tid  = threadIdx.x;
  const int wave = tid >> 5;
  const int lane = tid & 31;
  const int hl   = lane >> 4;
  const int l16  = lane & 15;
  const int m0   = blockIdx.x * 16;

  // --- cooperative stage: X[m0..m0+15][0..1023] f32 -> LDS f16 ---
  {
    const int r  = tid >> 4;            // one row per 16 threads
    const int c0 = (tid & 15) * 64;     // 64 columns each
    const float* p = X + (size_t)(m0 + r) * E_DIM + c0;
#pragma unroll
    for (int i = 0; i < 64; i += 8) {
      const float4 f0 = *(const float4*)(p + i);
      const float4 f1 = *(const float4*)(p + i + 4);
      v8h o;
      o[0] = (_Float16)f0.x; o[1] = (_Float16)f0.y;
      o[2] = (_Float16)f0.z; o[3] = (_Float16)f0.w;
      o[4] = (_Float16)f1.x; o[5] = (_Float16)f1.y;
      o[6] = (_Float16)f1.z; o[7] = (_Float16)f1.w;
      *(v8h*)(&xs[r][c0 + i]) = o;
    }
  }
  __syncthreads();

  const int n0 = wave * 128;            // this wave's 128 output columns
  v8f acc[8];
#pragma unroll
  for (int j = 0; j < 8; ++j) acc[j] = (v8f)0.0f;

  for (int e0 = 0; e0 < E_DIM; e0 += 32) {
    const v16h a = ld_afrag(&xs[l16][e0 + 8 * hl]);
#pragma unroll
    for (int j = 0; j < 8; ++j) {
      // B[k][n] = W[n][k]: W16 row n is contiguous in k
      const _Float16* p =
          W16 + (size_t)(n0 + j * 16 + l16) * E_DIM + e0 + 8 * hl;
      acc[j] = wmma_f16(a, ld_bfrag(p), acc[j]);
    }
  }

  // bias + store (C layout: lane l16 -> N col, VGPR v -> row v + 8*hl)
#pragma unroll
  for (int j = 0; j < 8; ++j) {
    const int col  = n0 + j * 16 + l16;
    const float bj = bias[col];
    if (transposeOut) {
      const int bb = m0 >> 11;
      const int s  = (m0 & (S_LEN - 1)) + 8 * hl;
      v8h pk;
#pragma unroll
      for (int v = 0; v < 8; ++v) pk[v] = (_Float16)(acc[j][v] + bj);
      *(v8h*)(out + ((size_t)bb * E_DIM + col) * S_LEN + s) = pk;
    } else {
#pragma unroll
      for (int v = 0; v < 8; ++v) {
        const int row = m0 + v + 8 * hl;
        out[(size_t)row * E_DIM + col] = (_Float16)(acc[j][v] + bj);
      }
    }
  }
}

// ---------------------------------------------------------------------------
// Kernel 2: fused flash attention.
// Block = (16 query rows, one batch), 256 threads = 8 waves.
// Wave w owns output e-slice [w*128, w*128+128).
// ---------------------------------------------------------------------------
__global__ __launch_bounds__(256) void attn_kernel(const _Float16* __restrict__ qf,
                                                   const _Float16* __restrict__ kf,
                                                   const _Float16* __restrict__ vT,
                                                   const float* __restrict__ mask,
                                                   float* __restrict__ out) {
  __shared__ float    s_part[8][16][TK];   // per-wave partial scores (32 KB)
  __shared__ float    s_red[16][TK];       // reduced scores
  __shared__ _Float16 p_lds[16][TK];       // softmax probs (A operand of P@V)
  __shared__ float    mlds[16][TK];        // async-staged mask block
  __shared__ float    rowM[16], rowL[16], rowScale[16];

  const int tid  = threadIdx.x;
  const int wave = tid >> 5;
  const int lane = tid & 31;
  const int hl   = lane >> 4;
  const int l16  = lane & 15;

  const int b  = blockIdx.y;
  const int m0 = blockIdx.x * 16;
  const int ew = wave * 128;

  const _Float16* Q  = qf + ((size_t)b * S_LEN + m0) * E_DIM;
  const _Float16* K  = kf + (size_t)b * S_LEN * E_DIM;
  const _Float16* Vt = vT + (size_t)b * E_DIM * S_LEN;           // [E][S]
  const float*    Mk = mask + (size_t)b * S_LEN * S_LEN + (size_t)m0 * S_LEN;

  // Preload Q A-fragments for this wave's e-slice (reused for all key blocks)
  v16h aq[4];
#pragma unroll
  for (int c = 0; c < 4; ++c)
    aq[c] = ld_afrag(Q + (size_t)l16 * E_DIM + ew + c * 32 + 8 * hl);

  v8f acc[8];
#pragma unroll
  for (int j = 0; j < 8; ++j) acc[j] = (v8f)0.0f;

  if (tid < 16) { rowM[tid] = -1e30f; rowL[tid] = 0.0f; }
  __syncthreads();

  for (int t0 = 0; t0 < S_LEN; t0 += TK) {
    // --- 0) async-stage this block's mask tile (16x64 f32) into LDS;
    //        overlaps with the Q*K^T WMMA phase below ---
    if (wave == 0) {
#pragma unroll
      for (int it = 0; it < 8; ++it) {
        const int idx = it * 32 + lane;       // 256 chunks of 16B
        const int r = idx >> 4, ci = idx & 15;
        async_g2l_b128((uint32_t)(uintptr_t)&mlds[r][ci * 4],
                       Mk + (size_t)r * S_LEN + t0 + ci * 4);
      }
    }
    if (t0 + TK < S_LEN)
      __builtin_prefetch(K + (size_t)(t0 + TK + l16) * E_DIM + ew, 0, 1);

    // --- 1) partial scores over this wave's e-slice: S_w = Q_w * K_w^T ---
    v8f sacc[4];
#pragma unroll
    for (int j = 0; j < 4; ++j) sacc[j] = (v8f)0.0f;
#pragma unroll
    for (int c = 0; c < 4; ++c) {
#pragma unroll
      for (int j = 0; j < 4; ++j) {
        const _Float16* p =
            K + (size_t)(t0 + j * 16 + l16) * E_DIM + ew + c * 32 + 8 * hl;
        sacc[j] = wmma_f16(aq[c], ld_bfrag(p), sacc[j]);
      }
    }
#pragma unroll
    for (int j = 0; j < 4; ++j)
#pragma unroll
      for (int v = 0; v < 8; ++v)
        s_part[wave][v + 8 * hl][j * 16 + l16] = sacc[j][v];
    if (wave == 0) wait_async0();     // mask tile landed before the barrier
    __syncthreads();

    // --- 2) reduce partials across the 8 waves ---
#pragma unroll
    for (int e = 0; e < 4; ++e) {
      const int idx = tid * 4 + e;
      const int r = idx >> 6, c = idx & 63;
      float s = 0.0f;
#pragma unroll
      for (int w = 0; w < 8; ++w) s += s_part[w][r][c];
      s_red[r][c] = s;
    }
    __syncthreads();

    // --- 3) online softmax, fully parallel: 16 threads per row, 4 cols each,
    //        shfl_xor reductions across the 16-lane row group ---
    {
      const int r  = tid >> 4;
      const int cg = tid & 15;
      const float rm_old = rowM[r];
      float vals[4];
      float mx = -3.0e38f;
#pragma unroll
      for (int i = 0; i < 4; ++i) {
        const int c = cg * 4 + i;
        const float sv = s_red[r][c] * 0.03125f + mlds[r][c];  // 1/sqrt(1024)
        vals[i] = sv;
        mx = fmaxf(mx, sv);
      }
#pragma unroll
      for (int m = 8; m >= 1; m >>= 1) mx = fmaxf(mx, __shfl_xor(mx, m, 16));
      const float mnew = fmaxf(rm_old, mx);
      float s = 0.0f;
#pragma unroll
      for (int i = 0; i < 4; ++i) {
        const float p = __expf(vals[i] - mnew);
        s += p;
        p_lds[r][cg * 4 + i] = (_Float16)p;
      }
#pragma unroll
      for (int m = 8; m >= 1; m >>= 1) s += __shfl_xor(s, m, 16);
      if (cg == 0) {
        const float sc = __expf(rm_old - mnew);
        rowScale[r] = sc;
        rowL[r] = rowL[r] * sc + s;
        rowM[r] = mnew;
      }
    }
    __syncthreads();

    // --- 4) rescale running output, then O += P @ V (e-slice) ---
    {
      float scv[8];
#pragma unroll
      for (int v = 0; v < 8; ++v) scv[v] = rowScale[v + 8 * hl];
#pragma unroll
      for (int j = 0; j < 8; ++j)
#pragma unroll
        for (int v = 0; v < 8; ++v) acc[j][v] *= scv[v];
    }
#pragma unroll
    for (int c2 = 0; c2 < 2; ++c2) {
      const v16h ap = ld_afrag(&p_lds[l16][c2 * 32 + 8 * hl]);
#pragma unroll
      for (int j = 0; j < 8; ++j) {
        // B[k][e] = V[t0+k][ew+j*16+n] = vT[ew+j*16+n][t0+k] (contiguous in k)
        const _Float16* p =
            Vt + (size_t)(ew + j * 16 + l16) * S_LEN + t0 + c2 * 32 + 8 * hl;
        acc[j] = wmma_f16(ap, ld_bfrag(p), acc[j]);
      }
    }
    __syncthreads();
  }

  // --- finalize: O /= l, store f32 ---
  float rl[8];
#pragma unroll
  for (int v = 0; v < 8; ++v) rl[v] = 1.0f / rowL[v + 8 * hl];
#pragma unroll
  for (int j = 0; j < 8; ++j) {
    const int col = ew + j * 16 + l16;
#pragma unroll
    for (int v = 0; v < 8; ++v) {
      const int row = m0 + v + 8 * hl;
      out[((size_t)b * S_LEN + row) * E_DIM + col] = acc[j][v] * rl[v];
    }
  }
}

// ---------------------------------------------------------------------------
extern "C" void kernel_launch(void* const* d_in, const int* in_sizes, int n_in,
                              void* d_out, int out_size, void* d_ws, size_t ws_size,
                              hipStream_t stream) {
  const float* query = (const float*)d_in[0];
  const float* key   = (const float*)d_in[1];
  const float* value = (const float*)d_in[2];
  const float* amask = (const float*)d_in[3];
  const float* Wq    = (const float*)d_in[4];
  const float* bq    = (const float*)d_in[5];
  const float* Wk    = (const float*)d_in[6];
  const float* bk    = (const float*)d_in[7];
  const float* Wv    = (const float*)d_in[8];
  const float* bv    = (const float*)d_in[9];

  const size_t tok  = (size_t)4 * S_LEN * E_DIM;   // 8M elements
  const int    wN   = E_DIM * E_DIM;               // 1M elements
  _Float16* qf  = (_Float16*)d_ws;
  _Float16* kf  = qf + tok;
  _Float16* vT  = kf + tok;
  _Float16* w16q = vT + tok;
  _Float16* w16k = w16q + wN;
  _Float16* w16v = w16k + wN;                      // total 54 MB scratch

  cvt16_kernel<<<512, 256, 0, stream>>>(Wq, w16q, wN);
  cvt16_kernel<<<512, 256, 0, stream>>>(Wk, w16k, wN);
  cvt16_kernel<<<512, 256, 0, stream>>>(Wv, w16v, wN);

  proj_kernel<<<512, 256, 0, stream>>>(query, w16q, bq, qf, 0);
  proj_kernel<<<512, 256, 0, stream>>>(key,   w16k, bk, kf, 0);
  proj_kernel<<<512, 256, 0, stream>>>(value, w16v, bv, vT, 1);

  attn_kernel<<<dim3(S_LEN / 16, 4), 256, 0, stream>>>(qf, kf, vT, amask,
                                                       (float*)d_out);
}
